// DualMixtureV2Mutator_13357348290793
// MI455X (gfx1250) — matrix operand order repaired and checked by reading, not compiled, not run on hardware
//
#include <hip/hip_runtime.h>
#include <hip/hip_bf16.h>

typedef __attribute__((ext_vector_type(16))) _Float16 v16h;
typedef __attribute__((ext_vector_type(8)))  _Float16 v8h;
typedef __attribute__((ext_vector_type(4)))  _Float16 v4h;
typedef __attribute__((ext_vector_type(8)))  float    v8f;

namespace {

constexpr int kCS = 64;
constexpr int kRD = 16;
constexpr int kH  = 64;
constexpr int kRANK = 4;
constexpr int kG = 8;
constexpr int kNPol = 8388608 / kCS;      // 131072 policy rows (== meta rows)
constexpr int kWaves = 4;                 // 4 wave32 per block
constexpr int kThreads = kWaves * 32;     // 128
constexpr int kTiles = 4;                 // 16-row tiles per wave
constexpr int kRowsPerBlk = kWaves * kTiles * 16;   // 256 rows / block
constexpr int kBlocks = kNPol / kRowsPerBlk;        // 512

// workspace float layout
constexpr int WS_GSUM  = 0;     // 8*16
constexpr int WS_GCNT  = 128;   // 8
constexpr int WS_CTX   = 136;   // 8*16
constexpr int WS_GSC   = 264;   // 8
constexpr int WS_BASIS = 272;   // 8*4*64

// branch-free tanh: 1 - 2/(e^{2x}+1); saturates to +-1, no divergent paths
__device__ __forceinline__ float fast_tanh(float x) {
  float e2 = __builtin_amdgcn_exp2f(x * 2.885390081777927f);  // e^(2x)
  return 1.0f - 2.0f * __builtin_amdgcn_rcpf(e2 + 1.0f);
}

// ---- WMMA fragment builders (documented CDNA5 wave32 layouts) ----
// A (16x32 f16, row-major LDS, ld halves): two aligned 16B loads per lane.
__device__ __forceinline__ v16h ldsA(const _Float16* s, int ld) {
  const int lane = threadIdx.x & 31;
  const _Float16* p = s + (lane & 15) * ld + ((lane >> 4) << 3);
  v8h a0 = *(const v8h*)p;
  v8h a1 = *(const v8h*)(p + 16);
  return __builtin_shufflevector(a0, a1, 0, 1, 2, 3, 4, 5, 6, 7,
                                 8, 9, 10, 11, 12, 13, 14, 15);
}

// B fragment pre-swizzled in LDS: lane's 16 halves contiguous -> 2x ds_load_b128
__device__ __forceinline__ v16h ldsBf(const _Float16* s) {
  return *(const v16h*)(s + ((threadIdx.x & 31) << 4));
}

// Stage one 32x16 B tile (K x N slice of row-major W, zero-padded) into
// fragment-major order: dst[lane*16 + k] = W[(krow0 + (k>>3)*16 + (lane>>4)*8 + (k&7))][ncol0 + (lane&15)]
__device__ __forceinline__ void stageBtile(_Float16* dst, const float* __restrict__ W,
                                           int N, int krow0, int ncol0,
                                           int kValid, int nTileValid) {
  for (int i = threadIdx.x; i < 512; i += kThreads) {
    int lane = i >> 4, k = i & 15;
    int kk = krow0 + ((k >> 3) << 4) + ((lane >> 4) << 3) + (k & 7);
    int nn = lane & 15;
    float v = (kk < kValid && nn < nTileValid) ? W[kk * N + ncol0 + nn] : 0.f;
    dst[i] = (_Float16)v;
  }
}

// C/D f32 16x16: row = v + 8*(lane>=16), col = lane&15
__device__ __forceinline__ v8f wmma16(v16h a, v16h b, v8f c) {
  return __builtin_amdgcn_wmma_f32_16x16x32_f16(
      false, a, false, b, (short)0, c, false, false);
}

__device__ __forceinline__ int groupOf(const long long* bnd, long long start) {
  int g = 0;
#pragma unroll
  for (int i = 0; i < kG - 1; ++i) g += (bnd[i] < start) ? 1 : 0;
  return g;  // == clip(searchsorted(bnd, start, 'left'), 0, 7)
}

// -------------------- kernel 0: zero group accumulators --------------------
__global__ void k_zero(float* __restrict__ ws) {
  int i = blockIdx.x * blockDim.x + threadIdx.x;
  if (i < kG * kRD + kG) ws[i] = 0.f;
}

// -------------------- kernel 1: encoder + group segment sums ---------------
__global__ __launch_bounds__(kThreads) void k_gsum(
    const float* __restrict__ flat, const float* __restrict__ encW,
    const float* __restrict__ encB, const long long* __restrict__ bnd,
    float* __restrict__ ws) {
  __shared__ __align__(32) _Float16 sFB[2 * 512];
  __shared__ float sEncB[kRD];
  __shared__ float sGS[kG][kRD];
  __shared__ float sCnt[kG];
  __shared__ long long sBnd[kG - 1];
  __shared__ __align__(16) _Float16 sAct[kWaves * 16 * kCS];
  __shared__ int sGrp[kWaves * 16];

  const int tid = threadIdx.x;
  stageBtile(sFB + 0 * 512, encW, kRD, 0, 0, 64, 16);
  stageBtile(sFB + 1 * 512, encW, kRD, 32, 0, 64, 16);
  if (tid < kRD) sEncB[tid] = encB[tid];
  if (tid < kG - 1) sBnd[tid] = bnd[tid];
  for (int i = tid; i < kG * kRD; i += kThreads) (&sGS[0][0])[i] = 0.f;
  if (tid < kG) sCnt[tid] = 0.f;
  __syncthreads();

  const int wave = tid >> 5, lane = tid & 31;
  const int ncol = lane & 15, hi = lane >> 4;
  _Float16* act = sAct + wave * 16 * kCS;
  int* grp = sGrp + wave * 16;

  for (int it = 0; it < kTiles; ++it) {
    const int row0 = blockIdx.x * kRowsPerBlk + (wave * kTiles + it) * 16;
    if (lane < 16) {
      int g = groupOf(sBnd, (long long)(row0 + lane) * kCS);
      grp[lane] = g;
      atomicAdd(&sCnt[g], 1.f);
    }
    const float4* xs = (const float4*)(flat + (size_t)row0 * kCS);
#pragma unroll
    for (int i = 0; i < 8; ++i) {
      int li = i * 32 + lane;
      int r = li >> 4, c4 = (li & 15) << 2;
      float4 v = xs[li];
      v4h h;
      h[0] = (_Float16)v.x; h[1] = (_Float16)v.y;
      h[2] = (_Float16)v.z; h[3] = (_Float16)v.w;
      *(v4h*)(act + r * kCS + c4) = h;
    }
    v8f acc = {};
    acc = wmma16(ldsA(act, kCS),      ldsBf(sFB + 0 * 512), acc);
    acc = wmma16(ldsA(act + 32, kCS), ldsBf(sFB + 1 * 512), acc);
#pragma unroll
    for (int v = 0; v < 8; ++v) {
      int r = hi * 8 + v;
      float e = fast_tanh(acc[v] + sEncB[ncol]);
      atomicAdd(&sGS[grp[r]][ncol], e);
    }
  }
  __syncthreads();
  if (tid < kG * kRD) atomicAdd(&ws[WS_GSUM + tid], (&sGS[0][0])[tid]);
  if (tid < kG) atomicAdd(&ws[WS_GCNT + tid], sCnt[tid]);
}

// -------------------- kernel 2: group ctx, scales, basis MLP ---------------
__global__ __launch_bounds__(128) void k_ctx(
    const float* __restrict__ refP, const float* __restrict__ baW1,
    const float* __restrict__ baB1, const float* __restrict__ baW2,
    const float* __restrict__ baB2, const float* __restrict__ gls,
    float* __restrict__ ws) {
  __shared__ float sCtx[kG][kRD];
  __shared__ float sH[kG][kH];
  const int tid = threadIdx.x;
  if (tid < kG * kRD) {
    int g = tid >> 4, c = tid & 15;
    float cnt = ws[WS_GCNT + g];
    float v = (cnt > 0.f) ? ws[WS_GSUM + tid] / fmaxf(cnt, 1.f) : refP[c];
    sCtx[g][c] = v;
    ws[WS_CTX + tid] = v;
  }
  if (tid < kG)
    ws[WS_GSC + tid] = __builtin_amdgcn_exp2f(gls[tid] * 1.4426950408889634f);
  __syncthreads();
  for (int i = tid; i < kG * kH; i += 128) {
    int g = i >> 6, j = i & 63;
    float s = baB1[j];
#pragma unroll
    for (int k = 0; k < kRD; ++k) s += sCtx[g][k] * baW1[k * kH + j];
    sH[g][j] = fast_tanh(s);
  }
  __syncthreads();
  for (int i = tid; i < kG * kRANK * kCS; i += 128) {
    int g = i >> 8, o = i & 255;
    float s = baB2[o];
    for (int j = 0; j < kH; ++j) s += sH[g][j] * baW2[j * (kRANK * kCS) + o];
    ws[WS_BASIS + i] = s;  // basis_g[g][r][c] at r*64+c
  }
}

// -------------------- kernel 3: policy rows --------------------------------
// sFB tile map: 0..1 enc | 2..5 co_W1 | 6..7 co_W2 | 8..15 pc_W1 | 16..23 pc_W2
constexpr int FB_ENC = 0, FB_C1 = 2, FB_C2 = 6, FB_W1 = 8, FB_W2 = 16;

__global__ __launch_bounds__(kThreads) void k_policy(
    const float* __restrict__ flat, const float* __restrict__ refP,
    const float* __restrict__ encW, const float* __restrict__ encB,
    const float* __restrict__ pcW1, const float* __restrict__ pcB1,
    const float* __restrict__ pcW2, const float* __restrict__ pcB2,
    const float* __restrict__ coW1, const float* __restrict__ coB1,
    const float* __restrict__ coW2, const float* __restrict__ coB2,
    const float* __restrict__ pcsP, const float* __restrict__ pnsP,
    const float* __restrict__ noise, const long long* __restrict__ bnd,
    const float* __restrict__ ws, float* __restrict__ out) {
  __shared__ __align__(32) _Float16 sFB[24 * 512];
  __shared__ float sEncB[kRD], sRef[kRD];
  __shared__ float sB1[kH], sB2[kCS], sCb1[kH], sCb2[kRANK];
  __shared__ float sCtx[kG][kRD];
  __shared__ float sBasis[kG][kRANK * kCS];
  __shared__ float sGsc[kG];
  __shared__ long long sBnd[kG - 1];
  __shared__ __align__(16) _Float16 sAct[kWaves * 16 * kCS];
  __shared__ __align__(16) _Float16 sHb[kWaves * 16 * kCS];
  __shared__ __align__(16) _Float16 sEnc[kWaves * 16 * kRD];
  __shared__ float sCoef[kWaves * 16 * kRANK];
  __shared__ int sGrp[kWaves * 16];

  const int tid = threadIdx.x;
  stageBtile(sFB + (FB_ENC + 0) * 512, encW, kRD, 0, 0, 64, 16);
  stageBtile(sFB + (FB_ENC + 1) * 512, encW, kRD, 32, 0, 64, 16);
#pragma unroll
  for (int t = 0; t < 4; ++t)
    stageBtile(sFB + (FB_C1 + t) * 512, coW1, kCS, 0, t * 16, 32, 16);
  stageBtile(sFB + (FB_C2 + 0) * 512, coW2, kRANK, 0, 0, 64, 4);
  stageBtile(sFB + (FB_C2 + 1) * 512, coW2, kRANK, 32, 0, 64, 4);
#pragma unroll
  for (int t = 0; t < 4; ++t)
#pragma unroll
    for (int kt = 0; kt < 2; ++kt) {
      stageBtile(sFB + (FB_W1 + t * 2 + kt) * 512, pcW1, kCS, kt * 32, t * 16, 48, 16);
      stageBtile(sFB + (FB_W2 + t * 2 + kt) * 512, pcW2, kCS, kt * 32, t * 16, 64, 16);
    }
  if (tid < kRD) { sEncB[tid] = encB[tid]; sRef[tid] = refP[tid]; }
  for (int i = tid; i < kH; i += kThreads) {
    sB1[i] = pcB1[i]; sB2[i] = pcB2[i]; sCb1[i] = coB1[i];
  }
  if (tid < kRANK) sCb2[tid] = coB2[tid];
  if (tid < kG) sGsc[tid] = ws[WS_GSC + tid];
  if (tid < kG * kRD) (&sCtx[0][0])[tid] = ws[WS_CTX + tid];
  for (int i = tid; i < kG * kRANK * kCS; i += kThreads)
    (&sBasis[0][0])[i] = ws[WS_BASIS + i];
  if (tid < kG - 1) sBnd[tid] = bnd[tid];
  __syncthreads();

  const float pcs  = *pcsP;
  const float pnsa = fabsf(*pnsP);

  const int wave = tid >> 5, lane = tid & 31;
  const int ncol = lane & 15, hi = lane >> 4;
  _Float16* act = sAct + wave * 16 * kCS;
  _Float16* hb  = sHb + wave * 16 * kCS;
  _Float16* enc = sEnc + wave * 16 * kRD;
  float* coef = sCoef + wave * 16 * kRANK;
  int* grp = sGrp + wave * 16;

  for (int it = 0; it < kTiles; ++it) {
    const int row0 = blockIdx.x * kRowsPerBlk + (wave * kTiles + it) * 16;
    const size_t base = (size_t)row0 * kCS;

    if (lane < 16) grp[lane] = groupOf(sBnd, (long long)(row0 + lane) * kCS);

    if (it + 1 < kTiles) {  // prefetch next tile's streams (global_prefetch_b8)
      __builtin_prefetch(flat + base + 16 * kCS + lane * 32, 0, 1);
      __builtin_prefetch(noise + base + 16 * kCS + lane * 32, 0, 1);
    }

    // X tile (16x64 f32 -> f16 LDS), packed 8B stores
    const float4* xs = (const float4*)(flat + base);
#pragma unroll
    for (int i = 0; i < 8; ++i) {
      int li = i * 32 + lane;
      int r = li >> 4, c4 = (li & 15) << 2;
      float4 v = xs[li];
      v4h h;
      h[0] = (_Float16)v.x; h[1] = (_Float16)v.y;
      h[2] = (_Float16)v.z; h[3] = (_Float16)v.w;
      *(v4h*)(act + r * kCS + c4) = h;
    }

    // encoder: (16x64)@(64x16)
    v8f eacc = {};
    eacc = wmma16(ldsA(act, kCS),      ldsBf(sFB + (FB_ENC + 0) * 512), eacc);
    eacc = wmma16(ldsA(act + 32, kCS), ldsBf(sFB + (FB_ENC + 1) * 512), eacc);
#pragma unroll
    for (int v = 0; v < 8; ++v) {
      int r = hi * 8 + v;
      enc[r * kRD + ncol] = (_Float16)fast_tanh(eacc[v] + sEncB[ncol]);
    }

    // coeff_in = [pe, ctx[g]] (16x32) into act
#pragma unroll
    for (int i = 0; i < 4; ++i) {
      int li = i * 32 + lane;
      int r = li >> 3, cb = (li & 7) << 2;
      v4h o;
      if (cb < 16) {
        o = *(const v4h*)(enc + r * kRD + cb);
      } else {
        const float* cx = sCtx[grp[r]] + (cb - 16);
        o[0] = (_Float16)cx[0]; o[1] = (_Float16)cx[1];
        o[2] = (_Float16)cx[2]; o[3] = (_Float16)cx[3];
      }
      *(v4h*)(act + r * kCS + cb) = o;
    }
    // coeff layer1: (16x32)@(32x64) -> tanh -> hb
#pragma unroll
    for (int t = 0; t < 4; ++t) {
      v8f c1 = {};
      c1 = wmma16(ldsA(act, kCS), ldsBf(sFB + (FB_C1 + t) * 512), c1);
      int cc = t * 16 + ncol;
#pragma unroll
      for (int v = 0; v < 8; ++v) {
        int r = hi * 8 + v;
        hb[r * kCS + cc] = (_Float16)fast_tanh(c1[v] + sCb1[cc]);
      }
    }
    // coeff layer2: (16x64)@(64x4 padded)
    v8f c2 = {};
    c2 = wmma16(ldsA(hb, kCS),      ldsBf(sFB + (FB_C2 + 0) * 512), c2);
    c2 = wmma16(ldsA(hb + 32, kCS), ldsBf(sFB + (FB_C2 + 1) * 512), c2);
    if (ncol < kRANK) {
#pragma unroll
      for (int v = 0; v < 8; ++v) {
        int r = hi * 8 + v;
        coef[r * kRANK + ncol] = c2[v] + sCb2[ncol];
      }
    }

    // corr_in = [pe, ref, pe-ref, 0pad] (16x64) into act
#pragma unroll
    for (int i = 0; i < 8; ++i) {
      int li = i * 32 + lane;
      int r = li >> 4, cb = (li & 15) << 2;
      v4h o;
      if (cb < 16) {
        o = *(const v4h*)(enc + r * kRD + cb);
      } else if (cb < 32) {
        const float* rp = sRef + (cb - 16);
        o[0] = (_Float16)rp[0]; o[1] = (_Float16)rp[1];
        o[2] = (_Float16)rp[2]; o[3] = (_Float16)rp[3];
      } else if (cb < 48) {
        v4h pe = *(const v4h*)(enc + r * kRD + (cb - 32));
        const float* rp = sRef + (cb - 32);
        o[0] = (_Float16)((float)pe[0] - rp[0]);
        o[1] = (_Float16)((float)pe[1] - rp[1]);
        o[2] = (_Float16)((float)pe[2] - rp[2]);
        o[3] = (_Float16)((float)pe[3] - rp[3]);
      } else {
        o[0] = o[1] = o[2] = o[3] = (_Float16)0.f;
      }
      *(v4h*)(act + r * kCS + cb) = o;
    }
    // corr layer1
#pragma unroll
    for (int t = 0; t < 4; ++t) {
      v8f h1 = {};
      h1 = wmma16(ldsA(act, kCS),      ldsBf(sFB + (FB_W1 + t * 2 + 0) * 512), h1);
      h1 = wmma16(ldsA(act + 32, kCS), ldsBf(sFB + (FB_W1 + t * 2 + 1) * 512), h1);
      int cc = t * 16 + ncol;
#pragma unroll
      for (int v = 0; v < 8; ++v) {
        int r = hi * 8 + v;
        hb[r * kCS + cc] = (_Float16)fast_tanh(h1[v] + sB1[cc]);
      }
    }
    // corr layer2 + fused low-rank / noise / clip / add-base epilogue
#pragma unroll
    for (int t = 0; t < 4; ++t) {
      v8f y = {};
      y = wmma16(ldsA(hb, kCS),      ldsBf(sFB + (FB_W2 + t * 2 + 0) * 512), y);
      y = wmma16(ldsA(hb + 32, kCS), ldsBf(sFB + (FB_W2 + t * 2 + 1) * 512), y);
      int cc = t * 16 + ncol;
#pragma unroll
      for (int v = 0; v < 8; ++v) {
        int r = hi * 8 + v;
        int g = grp[r];
        float lc = fast_tanh(y[v] + sB2[cc]);
        float lr = 0.f;
#pragma unroll
        for (int q = 0; q < kRANK; ++q)
          lr += coef[r * kRANK + q] * sBasis[g][q * kCS + cc];
        float gs = sGsc[g];
        size_t gi = base + (size_t)r * kCS + cc;
        float d = (lc * pcs + lr) * gs + noise[gi] * pnsa * gs;
        d = fminf(0.2f, fmaxf(-0.2f, d));
        out[gi] = flat[gi] + d;
      }
    }
  }
}

// -------------------- kernel 4: meta rows ----------------------------------
// sFB tile map: 0..1 enc | 2..9 mc_W1 | 10..17 mc_W2
constexpr int MB_ENC = 0, MB_W1 = 2, MB_W2 = 10;

__global__ __launch_bounds__(kThreads) void k_meta(
    const float* __restrict__ flat, const float* __restrict__ refM,
    const float* __restrict__ encW, const float* __restrict__ encB,
    const float* __restrict__ mcW1, const float* __restrict__ mcB1,
    const float* __restrict__ mcW2, const float* __restrict__ mcB2,
    const float* __restrict__ mcsP, const float* __restrict__ mnsP,
    const float* __restrict__ noise, float* __restrict__ out) {
  __shared__ __align__(32) _Float16 sFB[18 * 512];
  __shared__ float sEncB[kRD], sRef[kRD];
  __shared__ float sB1[kH], sB2[kCS];
  __shared__ __align__(16) _Float16 sAct[kWaves * 16 * kCS];
  __shared__ __align__(16) _Float16 sHb[kWaves * 16 * kCS];
  __shared__ __align__(16) _Float16 sEnc[kWaves * 16 * kRD];

  const int tid = threadIdx.x;
  stageBtile(sFB + (MB_ENC + 0) * 512, encW, kRD, 0, 0, 64, 16);
  stageBtile(sFB + (MB_ENC + 1) * 512, encW, kRD, 32, 0, 64, 16);
#pragma unroll
  for (int t = 0; t < 4; ++t)
#pragma unroll
    for (int kt = 0; kt < 2; ++kt) {
      stageBtile(sFB + (MB_W1 + t * 2 + kt) * 512, mcW1, kCS, kt * 32, t * 16, 48, 16);
      stageBtile(sFB + (MB_W2 + t * 2 + kt) * 512, mcW2, kCS, kt * 32, t * 16, 64, 16);
    }
  if (tid < kRD) { sEncB[tid] = encB[tid]; sRef[tid] = refM[tid]; }
  for (int i = tid; i < kH; i += kThreads) { sB1[i] = mcB1[i]; sB2[i] = mcB2[i]; }
  __syncthreads();

  const float mcs  = *mcsP;
  const float mnsa = fabsf(*mnsP);

  const int wave = tid >> 5, lane = tid & 31;
  const int ncol = lane & 15, hi = lane >> 4;
  _Float16* act = sAct + wave * 16 * kCS;
  _Float16* hb  = sHb + wave * 16 * kCS;
  _Float16* enc = sEnc + wave * 16 * kRD;

  for (int it = 0; it < kTiles; ++it) {
    const int row0 = blockIdx.x * kRowsPerBlk + (wave * kTiles + it) * 16;
    const size_t lbase = (size_t)row0 * kCS;              // meta-local
    const size_t gbase = ((size_t)kNPol * kCS) + lbase;   // global

    if (it + 1 < kTiles) {
      __builtin_prefetch(flat + gbase + 16 * kCS + lane * 32, 0, 1);
      __builtin_prefetch(noise + lbase + 16 * kCS + lane * 32, 0, 1);
    }

    const float4* xs = (const float4*)(flat + gbase);
#pragma unroll
    for (int i = 0; i < 8; ++i) {
      int li = i * 32 + lane;
      int r = li >> 4, c4 = (li & 15) << 2;
      float4 v = xs[li];
      v4h h;
      h[0] = (_Float16)v.x; h[1] = (_Float16)v.y;
      h[2] = (_Float16)v.z; h[3] = (_Float16)v.w;
      *(v4h*)(act + r * kCS + c4) = h;
    }

    v8f eacc = {};
    eacc = wmma16(ldsA(act, kCS),      ldsBf(sFB + (MB_ENC + 0) * 512), eacc);
    eacc = wmma16(ldsA(act + 32, kCS), ldsBf(sFB + (MB_ENC + 1) * 512), eacc);
#pragma unroll
    for (int v = 0; v < 8; ++v) {
      int r = hi * 8 + v;
      enc[r * kRD + ncol] = (_Float16)fast_tanh(eacc[v] + sEncB[ncol]);
    }

#pragma unroll
    for (int i = 0; i < 8; ++i) {
      int li = i * 32 + lane;
      int r = li >> 4, cb = (li & 15) << 2;
      v4h o;
      if (cb < 16) {
        o = *(const v4h*)(enc + r * kRD + cb);
      } else if (cb < 32) {
        const float* rp = sRef + (cb - 16);
        o[0] = (_Float16)rp[0]; o[1] = (_Float16)rp[1];
        o[2] = (_Float16)rp[2]; o[3] = (_Float16)rp[3];
      } else if (cb < 48) {
        v4h pe = *(const v4h*)(enc + r * kRD + (cb - 32));
        const float* rp = sRef + (cb - 32);
        o[0] = (_Float16)((float)pe[0] - rp[0]);
        o[1] = (_Float16)((float)pe[1] - rp[1]);
        o[2] = (_Float16)((float)pe[2] - rp[2]);
        o[3] = (_Float16)((float)pe[3] - rp[3]);
      } else {
        o[0] = o[1] = o[2] = o[3] = (_Float16)0.f;
      }
      *(v4h*)(act + r * kCS + cb) = o;
    }
#pragma unroll
    for (int t = 0; t < 4; ++t) {
      v8f h1 = {};
      h1 = wmma16(ldsA(act, kCS),      ldsBf(sFB + (MB_W1 + t * 2 + 0) * 512), h1);
      h1 = wmma16(ldsA(act + 32, kCS), ldsBf(sFB + (MB_W1 + t * 2 + 1) * 512), h1);
      int cc = t * 16 + ncol;
#pragma unroll
      for (int v = 0; v < 8; ++v) {
        int r = hi * 8 + v;
        hb[r * kCS + cc] = (_Float16)fast_tanh(h1[v] + sB1[cc]);
      }
    }
#pragma unroll
    for (int t = 0; t < 4; ++t) {
      v8f y = {};
      y = wmma16(ldsA(hb, kCS),      ldsBf(sFB + (MB_W2 + t * 2 + 0) * 512), y);
      y = wmma16(ldsA(hb + 32, kCS), ldsBf(sFB + (MB_W2 + t * 2 + 1) * 512), y);
      int cc = t * 16 + ncol;
#pragma unroll
      for (int v = 0; v < 8; ++v) {
        int r = hi * 8 + v;
        float mc = fast_tanh(y[v] + sB2[cc]);
        size_t li = lbase + (size_t)r * kCS + cc;
        size_t gi = gbase + (size_t)r * kCS + cc;
        float d = mc * mcs + noise[li] * mnsa;
        d = fminf(0.05f, fmaxf(-0.05f, d));
        out[gi] = flat[gi] + d;
      }
    }
  }
}

}  // namespace

extern "C" void kernel_launch(void* const* d_in, const int* in_sizes, int n_in,
                              void* d_out, int out_size, void* d_ws, size_t ws_size,
                              hipStream_t stream) {
  const float* flat = (const float*)d_in[0];
  const float* refP = (const float*)d_in[1];
  const float* refM = (const float*)d_in[2];
  const float* encW = (const float*)d_in[3];
  const float* encB = (const float*)d_in[4];
  const float* pcW1 = (const float*)d_in[5];
  const float* pcB1 = (const float*)d_in[6];
  const float* pcW2 = (const float*)d_in[7];
  const float* pcB2 = (const float*)d_in[8];
  const float* mcW1 = (const float*)d_in[9];
  const float* mcB1 = (const float*)d_in[10];
  const float* mcW2 = (const float*)d_in[11];
  const float* mcB2 = (const float*)d_in[12];
  const float* coW1 = (const float*)d_in[13];
  const float* coB1 = (const float*)d_in[14];
  const float* coW2 = (const float*)d_in[15];
  const float* coB2 = (const float*)d_in[16];
  const float* baW1 = (const float*)d_in[17];
  const float* baB1 = (const float*)d_in[18];
  const float* baW2 = (const float*)d_in[19];
  const float* baB2 = (const float*)d_in[20];
  const float* pcsP = (const float*)d_in[21];
  const float* pnsP = (const float*)d_in[22];
  const float* mcsP = (const float*)d_in[23];
  const float* mnsP = (const float*)d_in[24];
  const float* gls  = (const float*)d_in[25];
  const float* pNoise = (const float*)d_in[26];
  const float* mNoise = (const float*)d_in[27];
  const long long* bnd = (const long long*)d_in[28];
  (void)in_sizes; (void)n_in; (void)out_size; (void)ws_size;

  float* ws  = (float*)d_ws;
  float* out = (float*)d_out;

  k_zero<<<1, 256, 0, stream>>>(ws);
  k_gsum<<<kBlocks, kThreads, 0, stream>>>(flat, encW, encB, bnd, ws);
  k_ctx<<<1, 128, 0, stream>>>(refP, baW1, baB1, baW2, baB2, gls, ws);
  k_policy<<<kBlocks, kThreads, 0, stream>>>(
      flat, refP, encW, encB, pcW1, pcB1, pcW2, pcB2,
      coW1, coB1, coW2, coB2, pcsP, pnsP, pNoise, bnd, ws, out);
  k_meta<<<kBlocks, kThreads, 0, stream>>>(
      flat, refM, encW, encB, mcW1, mcB1, mcW2, mcB2,
      mcsP, mnsP, mNoise, out);
}
